// QauntMBBlock_82222853915064
// MI455X (gfx1250) — matrix operand order repaired and checked by reading, not compiled, not run on hardware
//
#include <hip/hip_runtime.h>
#include <hip/hip_bf16.h>

typedef __attribute__((ext_vector_type(16))) _Float16 v16h;
typedef __attribute__((ext_vector_type(8)))  float    v8f;
typedef __attribute__((ext_vector_type(2)))  float    v2f;

// Problem constants (from reference)
constexpr int NPIX = 1024;   // 32*32
constexpr int CIN  = 256;
constexpr int C3   = 768;    // 3*TOT

// ---------------------------------------------------------------------------
// Kernel 1: qkv = W_qkv[768x256] * x[b][256x1024]   (f16 WMMA, f32 accum)
// one wave -> 16x64 output strip (4 N-tiles share one A fragment per K-step)
// ---------------------------------------------------------------------------
__global__ void __launch_bounds__(256) k_qkv_gemm(const float* __restrict__ x,
                                                  const float* __restrict__ Wqkv,
                                                  float* __restrict__ qkv)
{
  const int lane = threadIdx.x & 31;
  const int wid  = blockIdx.x * 8 + (threadIdx.x >> 5);   // 16*48*16 waves
  const int ntg = wid & 15;             // group of 4 pixel tiles
  const int mt  = (wid >> 4) % 48;      // out-channel tile
  const int b   = wid / (16 * 48);
  const int m  = lane & 15;
  const int kh = lane >> 4;
  const int col0 = ntg * 64 + m;

  const float* wrow = Wqkv + (size_t)(mt * 16 + m) * CIN;  // A: row m of W
  const float* xb   = x + (size_t)b * CIN * NPIX + col0;   // B: columns of x

  v8f acc[4] = {{}, {}, {}, {}};
  #pragma unroll 1
  for (int k0 = 0; k0 < CIN; k0 += 32) {
    v16h a;
    #pragma unroll
    for (int e = 0; e < 16; ++e) {      // A 16x32 f16, contiguous k = 16*kh + e (same order as B)
      const int ka = k0 + kh * 16 + e;
      a[e] = (_Float16)wrow[ka];
    }
    v16h bb[4];
    #pragma unroll
    for (int t = 0; t < 4; ++t)
      #pragma unroll
      for (int e = 0; e < 16; ++e) {    // B 32x16 f16: rows striped across lanes
        const int kb = k0 + kh * 16 + e;
        bb[t][e] = (_Float16)xb[(size_t)kb * NPIX + 16 * t];
      }
    if (k0 + 32 < CIN)
      __builtin_prefetch(xb + (size_t)(k0 + 32 + kh * 16) * NPIX, 0, 1);
    #pragma unroll
    for (int t = 0; t < 4; ++t)         // 4 WMMAs share SRC0 (A reuse)
      acc[t] = __builtin_amdgcn_wmma_f32_16x16x32_f16(false, a, false, bb[t],
                                                      (short)0, acc[t], false, false);
  }
  float* dst = qkv + (size_t)b * C3 * NPIX + col0;
  #pragma unroll
  for (int t = 0; t < 4; ++t)
    #pragma unroll
    for (int r = 0; r < 8; ++r)         // D VGPR r -> row r + 8*kh
      dst[(size_t)(mt * 16 + kh * 8 + r) * NPIX + 16 * t] = acc[t][r];
}

// ---------------------------------------------------------------------------
// Kernel 2: depthwise 5x5 conv, pad 2 -- one block per (b,c) 32x32 plane,
// plane staged in LDS (1x global read instead of ~9x), taps from ds_load
// ---------------------------------------------------------------------------
__global__ void __launch_bounds__(256) k_dwconv(const float* __restrict__ qkv,
                                                const float* __restrict__ Wdw,
                                                float* __restrict__ aggdw)
{
  __shared__ float4 tile4[256];                  // 4 KB: the whole plane
  float* tile = (float*)tile4;
  const int t  = threadIdx.x;
  const int bc = blockIdx.x;                     // b*768 + c
  const int c  = bc % C3;

  const float* src = qkv + (size_t)bc * NPIX;
  tile4[t] = reinterpret_cast<const float4*>(src)[t];   // global_load_b128
  __syncthreads();

  const float* wp = Wdw + c * 25;                // uniform -> SMEM loads
  float w[25];
  #pragma unroll
  for (int i = 0; i < 25; ++i) w[i] = wp[i];

  float* dst = aggdw + (size_t)bc * NPIX;
  #pragma unroll
  for (int i = 0; i < 4; ++i) {
    const int p  = t + 256 * i;
    const int yy = p >> 5, xx = p & 31;
    float s = 0.f;
    #pragma unroll
    for (int dy = -2; dy <= 2; ++dy) {
      const int y = yy + dy;
      if ((unsigned)y < 32u) {
        #pragma unroll
        for (int dx = -2; dx <= 2; ++dx) {
          const int xq = xx + dx;
          if ((unsigned)xq < 32u)
            s = fmaf(w[(dy + 2) * 5 + (dx + 2)], tile[y * 32 + xq], s);
        }
      }
    }
    dst[p] = s;
  }
}

// ---------------------------------------------------------------------------
// Kernel 3: grouped pointwise (48 groups, 16->16) -- exact f32 WMMA 16x16x4,
// one wave -> 16x64 strip (A fragment shared by 4 N-tiles)
// ---------------------------------------------------------------------------
__global__ void __launch_bounds__(256) k_pwconv(const float* __restrict__ aggdw,
                                                const float* __restrict__ Wpw,
                                                float* __restrict__ agg)
{
  const int lane = threadIdx.x & 31;
  const int wid  = blockIdx.x * 8 + (threadIdx.x >> 5);   // 16*48*16 waves
  const int ntg = wid & 15;
  const int g   = (wid >> 4) % 48;
  const int b   = wid / (16 * 48);
  const int m  = lane & 15;
  const int kh = lane >> 4;
  const int col0 = ntg * 64 + m;
  const float* wrow = Wpw + (g * 16 + m) * 16;
  const float* src  = aggdw + ((size_t)b * C3 + g * 16) * NPIX + col0;

  v8f acc[4] = {{}, {}, {}, {}};
  #pragma unroll
  for (int k0 = 0; k0 < 16; k0 += 4) {
    v2f a;                               // A 16x4 f32: {K=2*kh, K=2*kh+1}
    a.x = wrow[k0 + 2 * kh];
    a.y = wrow[k0 + 2 * kh + 1];
    #pragma unroll
    for (int t = 0; t < 4; ++t) {
      v2f bbv;
      bbv.x = src[(size_t)(k0 + 2 * kh) * NPIX + 16 * t];
      bbv.y = src[(size_t)(k0 + 2 * kh + 1) * NPIX + 16 * t];
      acc[t] = __builtin_amdgcn_wmma_f32_16x16x4_f32(false, a, false, bbv,
                                                     (short)0, acc[t], false, false);
    }
  }
  float* dst = agg + ((size_t)b * C3 + g * 16) * NPIX + col0;
  #pragma unroll
  for (int t = 0; t < 4; ++t)
    #pragma unroll
    for (int r = 0; r < 8; ++r)
      dst[(size_t)(kh * 8 + r) * NPIX + 16 * t] = acc[t][r];
}

// ---------------------------------------------------------------------------
// Kernel 4a: kv[d][e] = sum_n relu(k[n,d]) * v1[n,e]  (f16 WMMA, K=n split 8x)
// ones-column B fragment produces kv[d][16] = sum_n k[n,d] in the same stream
// ---------------------------------------------------------------------------
__global__ void __launch_bounds__(256) k_attn_kv(const float* __restrict__ qkv,
                                                 const float* __restrict__ agg,
                                                 float* __restrict__ kvbuf)
{
  const int lane = threadIdx.x & 31;
  const int wid  = blockIdx.x * 8 + (threadIdx.x >> 5);  // 16*32*8 waves
  const int ks = wid & 7;
  const int h  = (wid >> 3) % 32;
  const int b  = wid / (8 * 32);
  const int kh  = lane >> 4;
  const int l15 = lane & 15;

  const float* src = (h < 16)
      ? qkv + ((size_t)b * C3 + h * 48) * NPIX
      : agg + ((size_t)b * C3 + (h - 16) * 48) * NPIX;
  const float* kp = src + (size_t)16 * NPIX;   // k channels
  const float* vp = src + (size_t)32 * NPIX;   // v channels

  v16h ones = {};
  if (l15 == 0) {
    #pragma unroll
    for (int e = 0; e < 16; ++e) ones[e] = (_Float16)1.0f;
  }

  v8f acc = {}, accs = {};
  const int nbase = ks * 128;
  #pragma unroll 1
  for (int n0 = nbase; n0 < nbase + 128; n0 += 32) {
    v16h a, bb;
    #pragma unroll
    for (int e = 0; e < 16; ++e) {      // A[d][n] = relu(k[n,d]); contiguous in n
      const int na = n0 + kh * 16 + e;
      a[e] = (_Float16)fmaxf(0.f, kp[(size_t)l15 * NPIX + na]);
    }
    #pragma unroll
    for (int e = 0; e < 16; ++e) {      // B[n][e] = v[n,e]
      const int nb = n0 + kh * 16 + e;
      bb[e] = (_Float16)vp[(size_t)l15 * NPIX + nb];
    }
    acc  = __builtin_amdgcn_wmma_f32_16x16x32_f16(false, a, false, bb,
                                                  (short)0, acc, false, false);
    accs = __builtin_amdgcn_wmma_f32_16x16x32_f16(false, a, false, ones,
                                                  (short)0, accs, false, false);
  }
  float* kvt = kvbuf + (size_t)(b * 32 + h) * 16 * 17;
  #pragma unroll
  for (int r = 0; r < 8; ++r) {
    atomicAdd(&kvt[(kh * 8 + r) * 17 + l15], acc[r]);
    if (l15 == 0) atomicAdd(&kvt[(kh * 8 + r) * 17 + 16], accs[r]);
  }
}

// ---------------------------------------------------------------------------
// Kernel 4b: o[n,e] = (q kv)[n,e] / (q . ksum)[n]   (exact f32 WMMA 16x16x4)
// ---------------------------------------------------------------------------
__global__ void __launch_bounds__(256) k_attn_o(const float* __restrict__ qkv,
                                                const float* __restrict__ agg,
                                                const float* __restrict__ kvbuf,
                                                float* __restrict__ obuf)
{
  const int lane = threadIdx.x & 31;
  const int wid  = blockIdx.x * 8 + (threadIdx.x >> 5);  // 16*32*64 waves
  const int nt = wid & 63;
  const int h  = (wid >> 6) % 32;
  const int b  = wid / (64 * 32);
  const int kh  = lane >> 4;
  const int l15 = lane & 15;

  const float* src = (h < 16)
      ? qkv + ((size_t)b * C3 + h * 48) * NPIX
      : agg + ((size_t)b * C3 + (h - 16) * 48) * NPIX;   // q = channels 0..15
  const float* kvt = kvbuf + (size_t)(b * 32 + h) * 16 * 17;

  float ksum[16];
  #pragma unroll
  for (int d = 0; d < 16; ++d) ksum[d] = kvt[d * 17 + 16];

  const int n_me = nt * 16 + l15;       // A row (M = pixel)
  v8f acc = {};
  #pragma unroll
  for (int k0 = 0; k0 < 16; k0 += 4) {
    v2f a, bbv;
    a.x   = fmaxf(0.f, src[(size_t)(k0 + 2 * kh) * NPIX + n_me]);
    a.y   = fmaxf(0.f, src[(size_t)(k0 + 2 * kh + 1) * NPIX + n_me]);
    bbv.x = kvt[(k0 + 2 * kh) * 17 + l15];
    bbv.y = kvt[(k0 + 2 * kh + 1) * 17 + l15];
    acc = __builtin_amdgcn_wmma_f32_16x16x4_f32(false, a, false, bbv,
                                                (short)0, acc, false, false);
  }
  float* dst = obuf + ((size_t)b * 512 + h * 16 + l15) * NPIX;
  #pragma unroll
  for (int r = 0; r < 8; ++r) {
    const int n = nt * 16 + kh * 8 + r;
    float den = 0.f;
    #pragma unroll
    for (int d = 0; d < 16; ++d)
      den = fmaf(fmaxf(0.f, src[(size_t)d * NPIX + n]), ksum[d], den);
    dst[n] = acc[r] / (den + 1e-15f);
  }
}

// ---------------------------------------------------------------------------
// Kernel 5: out = W_proj[256x512] * o[b][512x1024] + x   (f16 WMMA + residual)
// one wave -> 16x64 strip (4 N-tiles share one A fragment)
// ---------------------------------------------------------------------------
__global__ void __launch_bounds__(256) k_proj(const float* __restrict__ obuf,
                                              const float* __restrict__ Wproj,
                                              const float* __restrict__ x,
                                              float* __restrict__ out)
{
  const int lane = threadIdx.x & 31;
  const int wid  = blockIdx.x * 8 + (threadIdx.x >> 5);  // 16*16*16 waves
  const int ntg = wid & 15;
  const int mt  = (wid >> 4) % 16;
  const int b   = wid / (16 * 16);
  const int kh = lane >> 4;
  const int m  = lane & 15;
  const int col0 = ntg * 64 + m;
  const float* wrow = Wproj + (size_t)(mt * 16 + m) * 512;
  const float* ob   = obuf + (size_t)b * 512 * NPIX + col0;

  v8f acc[4] = {{}, {}, {}, {}};
  #pragma unroll 1
  for (int k0 = 0; k0 < 512; k0 += 32) {
    v16h a;
    #pragma unroll
    for (int e = 0; e < 16; ++e) {
      const int ka = k0 + kh * 16 + e;
      a[e] = (_Float16)wrow[ka];
    }
    v16h bb[4];
    #pragma unroll
    for (int t = 0; t < 4; ++t)
      #pragma unroll
      for (int e = 0; e < 16; ++e) {
        const int kb = k0 + kh * 16 + e;
        bb[t][e] = (_Float16)ob[(size_t)kb * NPIX + 16 * t];
      }
    if (k0 + 32 < 512)
      __builtin_prefetch(ob + (size_t)(k0 + 32 + kh * 16) * NPIX, 0, 1);
    #pragma unroll
    for (int t = 0; t < 4; ++t)
      acc[t] = __builtin_amdgcn_wmma_f32_16x16x32_f16(false, a, false, bb[t],
                                                      (short)0, acc[t], false, false);
  }
  const size_t base = ((size_t)b * CIN + mt * 16 + kh * 8) * NPIX + col0;
  #pragma unroll
  for (int t = 0; t < 4; ++t)
    #pragma unroll
    for (int r = 0; r < 8; ++r)
      out[base + (size_t)r * NPIX + 16 * t] =
          acc[t][r] + x[base + (size_t)r * NPIX + 16 * t];
}

// ---------------------------------------------------------------------------
extern "C" void kernel_launch(void* const* d_in, const int* in_sizes, int n_in,
                              void* d_out, int out_size, void* d_ws, size_t ws_size,
                              hipStream_t stream) {
  const float* x     = (const float*)d_in[0];
  const float* Wqkv  = (const float*)d_in[1];
  const float* Wdw   = (const float*)d_in[2];
  const float* Wpw   = (const float*)d_in[3];
  const float* Wproj = (const float*)d_in[4];
  float* out = (float*)d_out;

  float* ws = (float*)d_ws;
  const size_t big = (size_t)16 * C3 * NPIX;     // 12,582,912 floats
  float* qkv   = ws;
  float* aggdw = ws + big;
  float* agg   = ws + 2 * big;
  float* kvbuf = ws + 3 * big;                    // 16*32*16*17 floats
  float* obuf  = aggdw;                           // alias: aggdw dead after k_pwconv

  hipMemsetAsync(kvbuf, 0, (size_t)16 * 32 * 16 * 17 * sizeof(float), stream);

  k_qkv_gemm<<<1536, 256, 0, stream>>>(x, Wqkv, qkv);          // 12288 waves
  k_dwconv  <<<12288, 256, 0, stream>>>(qkv, Wdw, aggdw);      // 1 block / plane
  k_pwconv  <<<1536, 256, 0, stream>>>(aggdw, Wpw, agg);       // 12288 waves
  k_attn_kv <<<512, 256, 0, stream>>>(qkv, agg, kvbuf);        // 4096 waves
  k_attn_o  <<<4096, 256, 0, stream>>>(qkv, agg, kvbuf, obuf); // 32768 waves
  k_proj    <<<512, 256, 0, stream>>>(obuf, Wproj, x, out);    // 4096 waves
}
